// CausalMultiHeadSelfAttention_42125039239930
// MI455X (gfx1250) — matrix operand order repaired and checked by reading, not compiled, not run on hardware
//
#include <hip/hip_runtime.h>
#include <hip/hip_bf16.h>

typedef __attribute__((ext_vector_type(16))) __bf16 v16bf;
typedef __attribute__((ext_vector_type(8)))  __bf16 v8bf;
typedef __attribute__((ext_vector_type(8)))  float  v8f;

#define WMMA_BF16(a, b, c) \
  __builtin_amdgcn_wmma_f32_16x16x32_bf16(false, (a), false, (b), (short)0, (c), false, false)

// ---- A-fragment loader (16-bit A 16x32 layout, ISA 7.12.2):
// lane half h holds K = {h*8 .. h*8+7} U {16+h*8 .. 16+h*8+7} of its row.
static __device__ __forceinline__ v16bf load_afrag(const __bf16* rowk, int half) {
    v8bf lo = *(const v8bf*)(rowk + half * 8);
    v8bf hi = *(const v8bf*)(rowk + 16 + half * 8);
    v16bf r;
#pragma unroll
    for (int i = 0; i < 8; ++i) { r[i] = lo[i]; r[i + 8] = hi[i]; }
    return r;
}

// ---- B-fragment: lane n (0-15) = column n; lane half h holds
// K = {h*8 .. h*8+7} U {16+h*8 .. 16+h*8+7} (same order as A).
static __device__ __forceinline__ v16bf load_bfrag(const __bf16* colk, int half) {
    v8bf lo = *(const v8bf*)(colk + half * 8);
    v8bf hi = *(const v8bf*)(colk + 16 + half * 8);
    v16bf r;
#pragma unroll
    for (int i = 0; i < 8; ++i) { r[i] = lo[i]; r[i + 8] = hi[i]; }
    return r;
}

// ======================= Tensor Data Mover (CDNA5) ==========================
#if __has_builtin(__builtin_amdgcn_tensor_load_to_lds) && \
    __has_builtin(__builtin_amdgcn_s_wait_tensorcnt)
#define HAVE_TDM 1
typedef __attribute__((ext_vector_type(4))) unsigned int u32x4;
typedef __attribute__((ext_vector_type(8))) int i32x8;
typedef __attribute__((ext_vector_type(4))) int i32x4;

// 2-D tile load: tile_h rows of tile_w bf16 elements, row stride stride_elems,
// from global gsrc into LDS byte address lds_addr. D# per ISA 8.3/8.4.
static __device__ __forceinline__ void tdm_load_2d(unsigned lds_addr, const void* gsrc,
                                                   unsigned tile_w, unsigned tile_h,
                                                   unsigned tensor_w, unsigned stride_elems) {
    unsigned long long ga = (unsigned long long)(size_t)gsrc;
    u32x4 g0;
    g0[0] = 1u;                                                  // count=1, user mode
    g0[1] = lds_addr;                                            // LDS byte address
    g0[2] = (unsigned)ga;                                        // global addr [31:0]
    g0[3] = (unsigned)((ga >> 32) & 0x01ffffffu) | (2u << 30);   // addr[56:32] | type=2
    i32x8 g1;
    g1[0] = (int)(1u << 16);                                     // data_size code 1 = 2B
    g1[1] = (int)((tensor_w & 0xffffu) << 16);                   // tensor_dim0 [15:0]
    g1[2] = (int)(((tensor_w >> 16) & 0xffffu) | (0xffffu << 16)); // dim0 hi | tensor_dim1 lo
    g1[3] = (int)((tile_w & 0xffffu) << 16);                     // tensor_dim1 hi=0 | tile_dim0
    g1[4] = (int)(tile_h & 0xffffu);                             // tile_dim1 (tile_dim2=0)
    g1[5] = (int)stride_elems;                                   // tensor_dim0_stride [31:0]
    g1[6] = 0;
    g1[7] = 0;
    i32x4 z4 = {0, 0, 0, 0};
    i32x8 z8 = {0, 0, 0, 0, 0, 0, 0, 0};
    // 6-arg form (this toolchain): (g0, g1, g2, g3, g4, cpol)
    __builtin_amdgcn_tensor_load_to_lds(g0, g1, z4, z4, z8, 0);
}
#else
// Fallback: cooperative vector copy (rows x row_elems bf16, row_elems % 8 == 0)
static __device__ __forceinline__ void stage_tile(__bf16* dst, const __bf16* src,
                                                  int rows, int row_elems, long src_stride,
                                                  int tid, int nthr) {
    int chunks = rows * (row_elems >> 3);
    for (int c = tid; c < chunks; c += nthr) {
        int r = c / (row_elems >> 3);
        int cc = (c % (row_elems >> 3)) << 3;
        *(v8bf*)(dst + r * row_elems + cc) = *(const v8bf*)(src + r * src_stride + cc);
    }
}
#endif

// ============================ cast fp32 -> bf16 =============================
__global__ __launch_bounds__(256)
void cast_f32_to_bf16(const float* __restrict__ src, __bf16* __restrict__ dst, long n) {
    long i = (long)blockIdx.x * blockDim.x + threadIdx.x;
    long stride = (long)gridDim.x * blockDim.x;
    for (; i < n; i += stride) dst[i] = (__bf16)src[i];
}

// =================== GEMM1: qkv = x @ Wqkv^T, scatter Q/K/V^T ===============
// Block: 256 thr = 8 waves (2 M x 4 N), tile 128x128, wave tile 64x32.
__global__ __launch_bounds__(256)
void gemm_qkv(const __bf16* __restrict__ A, const __bf16* __restrict__ B,
              __bf16* __restrict__ Qo, __bf16* __restrict__ Ko, __bf16* __restrict__ Vt) {
    const int lane = threadIdx.x & 31;
    const int wave = threadIdx.x >> 5;
    const int wm = wave >> 2, wn = wave & 3;
    const int half = lane >> 4, l16 = lane & 15;
    const long K = 1024;
    const long mBase = (long)blockIdx.x * 128 + wm * 64;
    const long nBase = (long)blockIdx.y * 128 + wn * 32;

    v8f acc[4][2] = {};
    for (int kk = 0; kk < 1024; kk += 32) {
        v16bf a[4], b[2];
#pragma unroll
        for (int i = 0; i < 4; ++i)
            a[i] = load_afrag(A + (mBase + i * 16 + l16) * K + kk, half);
#pragma unroll
        for (int j = 0; j < 2; ++j)
            b[j] = load_bfrag(B + (nBase + j * 16 + l16) * K + kk, half);
#pragma unroll
        for (int i = 0; i < 4; ++i)
#pragma unroll
            for (int j = 0; j < 2; ++j)
                acc[i][j] = WMMA_BF16(a[i], b[j], acc[i][j]);
    }

    const float qscale = 0.125f;  // 1/sqrt(64)
#pragma unroll
    for (int i = 0; i < 4; ++i) {
        const long m0 = mBase + i * 16;         // uniform; 16-row tile never crosses b
        const long bb = m0 >> 12;
        const long ss0 = (m0 & 4095) + half * 8;
#pragma unroll
        for (int j = 0; j < 2; ++j) {
            const long n0 = nBase + j * 16;     // uniform within tile
            const int sec = (int)(n0 >> 10);    // uniform -> scalar branch
            const long e0 = n0 & 1023;
            const long hd = e0 >> 6;            // uniform (16-tile inside one head)
            const long dd = (e0 & 63) + l16;
            const long bh = bb * 16 + hd;
            if (sec == 0) {
                __bf16* p = Qo + (bh * 4096 + ss0) * 64 + dd;
#pragma unroll
                for (int r = 0; r < 8; ++r) p[r * 64] = (__bf16)(acc[i][j][r] * qscale);
            } else if (sec == 1) {
                __bf16* p = Ko + (bh * 4096 + ss0) * 64 + dd;
#pragma unroll
                for (int r = 0; r < 8; ++r) p[r * 64] = (__bf16)acc[i][j][r];
            } else {
                // V^T: consecutive along r -> one packed 16B store
                __bf16* p = Vt + (bh * 64 + dd) * 4096 + ss0;
                v8bf pk;
#pragma unroll
                for (int r = 0; r < 8; ++r) pk[r] = (__bf16)acc[i][j][r];
                *(v8bf*)p = pk;
            }
        }
    }
}

// ======================= Flash attention (causal) ===========================
// Q,K: [bh, s, 64] bf16 (Q pre-scaled). Vt: [bh, 64, s] bf16.
// Block: 128 thr = 4 waves; 64 query rows per block (16 per wave).
// K/V tiles staged to LDS once per block via TDM (tensor_load_to_lds).
__global__ __launch_bounds__(128)
void flash_attn(const __bf16* __restrict__ Q, const __bf16* __restrict__ K,
                const __bf16* __restrict__ Vt, __bf16* __restrict__ O) {
    __shared__ __attribute__((aligned(32))) __bf16 kTile[64 * 64];   // [key][dh]
    __shared__ __attribute__((aligned(32))) __bf16 vTile[64 * 64];   // [dh][key]
    __shared__ __attribute__((aligned(32))) __bf16 plds[4][16 * 64]; // per-wave P
    const int lane = threadIdx.x & 31;
    const int wave = threadIdx.x >> 5;
    const int half = lane >> 4, l16 = lane & 15;
    const int qt = blockIdx.x;
    const int bh = blockIdx.y;
    const long S = 4096, DH = 64;
    const __bf16* Qh = Q + (long)bh * S * DH;
    const __bf16* Kh = K + (long)bh * S * DH;
    const __bf16* Vh = Vt + (long)bh * DH * S;
    const int q0 = qt * 64 + wave * 16;

    v16bf qf[2];
#pragma unroll
    for (int f = 0; f < 2; ++f)
        qf[f] = load_afrag(Qh + (long)(q0 + l16) * DH + f * 32, half);

#ifdef HAVE_TDM
    const unsigned kLds = (unsigned)(size_t)(void*)kTile;
    const unsigned vLds = (unsigned)(size_t)(void*)vTile;
#endif

    v8f accO[4] = {};
    float mrun[8], lrun[8];
#pragma unroll
    for (int r = 0; r < 8; ++r) { mrun[r] = -INFINITY; lrun[r] = 0.f; }

    for (int kt = 0; kt <= qt; ++kt) {
        const int k0 = kt * 64;
        // ---- stage K (64 keys x 64 dh) and V^T (64 dh x 64 keys) into LDS
#ifdef HAVE_TDM
        if (wave == 0) {
            tdm_load_2d(kLds, Kh + (long)k0 * DH, 64, 64, 64, 64);
            tdm_load_2d(vLds, Vh + k0, 64, 64, 4096, 4096);
            __builtin_amdgcn_s_wait_tensorcnt(0);
        }
#else
        stage_tile(kTile, Kh + (long)k0 * DH, 64, 64, 64, threadIdx.x, 128);
        stage_tile(vTile, Vh + k0, 64, 64, 4096, threadIdx.x, 128);
#endif
        __syncthreads();

        // ---- S = Q x K^T : 16x64 per wave (4 col tiles x 2 K-chunks)
        v8f s[4] = {};
#pragma unroll
        for (int j = 0; j < 4; ++j) {
            const __bf16* krow = kTile + (j * 16 + l16) * 64;
            s[j] = WMMA_BF16(qf[0], load_bfrag(krow, half), s[j]);
            s[j] = WMMA_BF16(qf[1], load_bfrag(krow + 32, half), s[j]);
        }
        if (kt == qt) {  // causal mask on diagonal tile only
#pragma unroll
            for (int j = 0; j < 4; ++j)
#pragma unroll
                for (int r = 0; r < 8; ++r) {
                    int row = q0 + r + half * 8;
                    int col = k0 + j * 16 + l16;
                    if (col > row) s[j][r] = -INFINITY;
                }
        }
        // ---- online softmax: each row lives in one 16-lane half, VGPR idx r
#pragma unroll
        for (int r = 0; r < 8; ++r) {
            float mt = fmaxf(fmaxf(s[0][r], s[1][r]), fmaxf(s[2][r], s[3][r]));
#pragma unroll
            for (int d = 1; d < 16; d <<= 1) mt = fmaxf(mt, __shfl_xor(mt, d, 32));
            float mnew = fmaxf(mrun[r], mt);
            float corr = __expf(mrun[r] - mnew);
            float psum = 0.f;
#pragma unroll
            for (int j = 0; j < 4; ++j) {
                float p = __expf(s[j][r] - mnew);
                s[j][r] = p;
                psum += p;
            }
#pragma unroll
            for (int d = 1; d < 16; d <<= 1) psum += __shfl_xor(psum, d, 32);
            lrun[r] = lrun[r] * corr + psum;
            mrun[r] = mnew;
#pragma unroll
            for (int j = 0; j < 4; ++j) accO[j][r] *= corr;
        }
        // ---- P (C-layout) -> LDS -> A-fragments
#pragma unroll
        for (int j = 0; j < 4; ++j)
#pragma unroll
            for (int r = 0; r < 8; ++r)
                plds[wave][(r + half * 8) * 64 + j * 16 + l16] = (__bf16)s[j][r];
        __syncthreads();
        // ---- O += P x V
#pragma unroll
        for (int f = 0; f < 2; ++f) {
            v16bf pf = load_afrag(&plds[wave][l16 * 64 + f * 32], half);
#pragma unroll
            for (int j = 0; j < 4; ++j) {
                v16bf vf = load_bfrag(vTile + (j * 16 + l16) * 64 + f * 32, half);
                accO[j] = WMMA_BF16(pf, vf, accO[j]);
            }
        }
        __syncthreads();  // protect kTile/vTile/plds before next iteration
    }

    // ---- normalize + write merged-head layout [b, s, h*64+dd] as bf16
    const int b = bh >> 4, h = bh & 15;
#pragma unroll
    for (int j = 0; j < 4; ++j) {
        __bf16* p = O + ((long)b * 4096 + q0 + half * 8) * 1024 + (long)h * 64 + j * 16 + l16;
#pragma unroll
        for (int r = 0; r < 8; ++r)
            p[(long)r * 1024] = (__bf16)(accO[j][r] / lrun[r]);
    }
}

// =================== GEMM2: out = attnout @ Wout^T (fp32 out) ===============
__global__ __launch_bounds__(256)
void gemm_out(const __bf16* __restrict__ A, const __bf16* __restrict__ B,
              float* __restrict__ C) {
    const int lane = threadIdx.x & 31;
    const int wave = threadIdx.x >> 5;
    const int wm = wave >> 2, wn = wave & 3;
    const int half = lane >> 4, l16 = lane & 15;
    const long K = 1024, N = 1024;
    const long mBase = (long)blockIdx.x * 128 + wm * 64;
    const long nBase = (long)blockIdx.y * 128 + wn * 32;

    v8f acc[4][2] = {};
    for (int kk = 0; kk < 1024; kk += 32) {
        v16bf a[4], b[2];
#pragma unroll
        for (int i = 0; i < 4; ++i)
            a[i] = load_afrag(A + (mBase + i * 16 + l16) * K + kk, half);
#pragma unroll
        for (int j = 0; j < 2; ++j)
            b[j] = load_bfrag(B + (nBase + j * 16 + l16) * K + kk, half);
#pragma unroll
        for (int i = 0; i < 4; ++i)
#pragma unroll
            for (int j = 0; j < 2; ++j)
                acc[i][j] = WMMA_BF16(a[i], b[j], acc[i][j]);
    }
#pragma unroll
    for (int i = 0; i < 4; ++i)
#pragma unroll
        for (int j = 0; j < 2; ++j) {
            float* p = C + (mBase + i * 16 + half * 8) * N + nBase + j * 16 + l16;
#pragma unroll
            for (int r = 0; r < 8; ++r) p[(long)r * N] = acc[i][j][r];
        }
}

// ================================ launcher ==================================
extern "C" void kernel_launch(void* const* d_in, const int* in_sizes, int n_in,
                              void* d_out, int out_size, void* d_ws, size_t ws_size,
                              hipStream_t stream) {
    const float* x    = (const float*)d_in[0];   // [2,4096,1024]
    const float* Wqkv = (const float*)d_in[1];   // [3072,1024]
    const float* Wout = (const float*)d_in[2];   // [1024,1024]
    float* out = (float*)d_out;                  // [2,4096,1024]

    const long NX = 8192L * 1024;
    const long NW1 = 3072L * 1024;
    const long NW2 = 1024L * 1024;

    char* ws = (char*)d_ws;
    size_t off = 0;
    auto take = [&](size_t bytes) -> void* {
        void* p = ws + off;
        off += (bytes + 255) & ~(size_t)255;
        return p;
    };
    __bf16* xb      = (__bf16*)take(NX * 2);
    __bf16* wqkvb   = (__bf16*)take(NW1 * 2);
    __bf16* woutb   = (__bf16*)take(NW2 * 2);
    __bf16* qb      = (__bf16*)take(NX * 2);   // [bh, s, 64]
    __bf16* kb      = (__bf16*)take(NX * 2);   // [bh, s, 64]
    __bf16* vtb     = (__bf16*)take(NX * 2);   // [bh, 64, s]
    __bf16* attnout = (__bf16*)take(NX * 2);   // [b, s, 1024]

    cast_f32_to_bf16<<<2048, 256, 0, stream>>>(x, xb, NX);
    cast_f32_to_bf16<<<1024, 256, 0, stream>>>(Wqkv, wqkvb, NW1);
    cast_f32_to_bf16<<<512, 256, 0, stream>>>(Wout, woutb, NW2);

    gemm_qkv<<<dim3(64, 24), 256, 0, stream>>>(xb, wqkvb, qb, kb, vtb);

    flash_attn<<<dim3(64, 32), 128, 0, stream>>>(qb, kb, vtb, attnout);

    gemm_out<<<dim3(64, 8), 256, 0, stream>>>(attnout, woutb, out);
}
